// CausalAttention_42726334660805
// MI455X (gfx1250) — matrix operand order repaired and checked by reading, not compiled, not run on hardware
//
#include <hip/hip_runtime.h>

#define BATCH 4
#define SEQ   4096
#define DIN   1024
#define DOUT  1024

// softmax scale folded with log2(e) so we can use exp2 everywhere
#define KSCALE_LOG2 (0.03125f * 1.44269504088896f)   // (1/sqrt(1024)) * log2(e)

typedef __attribute__((ext_vector_type(16))) __bf16         bf16x16;
typedef __attribute__((ext_vector_type(8)))  float          f32x8;
typedef __attribute__((ext_vector_type(8)))  unsigned short u16x8;

union Frag16 { bf16x16 v; u16x8 h[2]; };

// ---------------------------------------------------------------------------
// WMMA wrapper: D(f32 16x16) = A(bf16 16x32) * B(bf16 32x16) + C
// ---------------------------------------------------------------------------
__device__ __forceinline__ f32x8 wmma_bf16(bf16x16 a, bf16x16 b, f32x8 c) {
  return __builtin_amdgcn_wmma_f32_16x16x32_bf16(
      /*neg_a=*/false, a, /*neg_b=*/false, b,
      /*c_mod=*/(short)0, c, /*reuse_a=*/false, /*reuse_b=*/false);
}

// A fragment (16x32 bf16, M x K). p = tile origin, ld = row stride (elements).
// Lane 0-15: row=lane, K = {0..7, 16..23}; lane 16-31: row=lane-16, K = {8..15, 24..31}.
__device__ __forceinline__ bf16x16 load_frag_a(const __bf16* __restrict__ p,
                                               int ld, int lane) {
  int row = lane & 15;
  int k0  = (lane >> 4) * 8;
  const __bf16* q = p + (size_t)row * ld + k0;
  Frag16 f;
  f.h[0] = *(const u16x8*)(q);
  f.h[1] = *(const u16x8*)(q + 16);
  return f.v;
}
// Same, but source is LDS.
__device__ __forceinline__ bf16x16 load_frag_a_lds(const __bf16* p, int ld, int lane) {
  int row = lane & 15;
  int k0  = (lane >> 4) * 8;
  const __bf16* q = p + row * ld + k0;
  Frag16 f;
  f.h[0] = *(const u16x8*)(q);
  f.h[1] = *(const u16x8*)(q + 16);
  return f.v;
}

// B fragment (32x16 bf16, K x N), columns contiguous in memory:
// column n lives at p + n*ld, K ascending. Lane l: col = l&15, K = {0..7, 16..23} + (l>>4)*8.
__device__ __forceinline__ bf16x16 load_frag_b(const __bf16* __restrict__ p,
                                               int ld, int lane) {
  int col = lane & 15;
  int k0  = (lane >> 4) * 8;
  const __bf16* q = p + (size_t)col * ld + k0;
  Frag16 f;
  f.h[0] = *(const u16x8*)(q);
  f.h[1] = *(const u16x8*)(q + 16);
  return f.v;
}

// ---------------------------------------------------------------------------
// Phase 0: fp32 -> bf16 conversion
// ---------------------------------------------------------------------------
__global__ void cvt_f32_to_bf16(const float* __restrict__ in,
                                __bf16* __restrict__ out, int n) {
  int i      = blockIdx.x * blockDim.x + threadIdx.x;
  int stride = gridDim.x * blockDim.x;
  for (; i < n; i += stride) out[i] = (__bf16)in[i];
}

// ---------------------------------------------------------------------------
// Phase 1: projection GEMM  Y[r,e] = sum_k X[r,k] * W[e,k]
// X: [BATCH*SEQ, DIN] bf16 row-major. W: [DOUT, DIN] bf16 row-major
// (so B-fragment column e == W row e, contiguous).
// mode 0: store Y row-major bf16 [BATCH*SEQ, DOUT]  (Q, K)
// mode 1: store Y transposed per batch: Yt[b][e][n]  (V)
// One 16x16 tile per wave; 4 waves per block.
// ---------------------------------------------------------------------------
__global__ __launch_bounds__(128) void proj_gemm(
    const __bf16* __restrict__ X, const __bf16* __restrict__ W,
    __bf16* __restrict__ Yr, __bf16* __restrict__ Yt, int transposed) {
  int wid  = threadIdx.x >> 5;
  int lane = threadIdx.x & 31;
  int t    = blockIdx.x * 4 + wid;            // tile id
  int tr   = t >> 6;                          // row tile   (BATCH*SEQ/16 = 1024)
  int tc   = t & 63;                          // col tile   (DOUT/16 = 64)

  const __bf16* A  = X + (size_t)tr * 16 * DIN;
  const __bf16* Bp = W + (size_t)tc * 16 * DIN;

  f32x8 acc = {};
  #pragma unroll 4
  for (int k = 0; k < DIN; k += 32) {
    bf16x16 a = load_frag_a(A + k, DIN, lane);
    bf16x16 b = load_frag_b(Bp + k, DIN, lane);
    acc = wmma_bf16(a, b, acc);
  }

  int hi = lane >> 4, lo = lane & 15;
  if (!transposed) {
    #pragma unroll
    for (int r = 0; r < 8; ++r) {
      int m = 8 * hi + r;
      Yr[(size_t)(tr * 16 + m) * DOUT + tc * 16 + lo] = (__bf16)acc[r];
    }
  } else {
    int b  = (tr * 16) / SEQ;
    int nl = (tr * 16) % SEQ;
    __bf16* Vb = Yt + (size_t)b * DOUT * SEQ;
    #pragma unroll
    for (int r = 0; r < 8; ++r) {
      int m = 8 * hi + r;
      Vb[(size_t)(tc * 16 + lo) * SEQ + nl + m] = (__bf16)acc[r];
    }
  }
}

// ---------------------------------------------------------------------------
// Phase 2: causal flash attention.
// One 128-thread block (4 waves) per 16-row query tile.
// Per iteration the block consumes a 128-key super-block:
//   wave w computes S for keys [it*128 + 32w, +32)          (64 wmma)
//   softmax stats exchanged via LDS; P (16x128 bf16) staged in LDS
//   wave w accumulates O for D columns [256w, 256w+256) in REGISTERS
//   (16 x f32x8 = 128 VGPRs), alpha-rescale folded in        (64 wmma)
// Qb, Kb: [BATCH*SEQ, DOUT] bf16 row-major. Vt: [BATCH][DOUT][SEQ] bf16.
// ---------------------------------------------------------------------------
__global__ __launch_bounds__(128) void attn_fwd(
    const __bf16* __restrict__ Qb, const __bf16* __restrict__ Kb,
    const __bf16* __restrict__ Vt, float* __restrict__ out) {
  __shared__ __align__(16) __bf16 Qlds[16 * DOUT];   // 32 KB staged Q tile
  __shared__ __align__(16) __bf16 Plds[16 * 128];    // 4 KB  staged P tile
  __shared__ float statsM[4][16];
  __shared__ float statsS[4][16];

  int tile = blockIdx.x;                 // 0 .. BATCH*SEQ/16 - 1
  int b    = tile >> 8;                  // tile / (SEQ/16)
  int i0   = (tile & 255) * 16;          // first query row of this tile
  int tid  = threadIdx.x;
  int w    = tid >> 5;                   // wave id 0..3
  int lane = tid & 31;
  int hi   = lane >> 4, lo = lane & 15;

  const __bf16* Q = Qb + ((size_t)b * SEQ + i0) * DOUT;
  const __bf16* K = Kb + (size_t)b * SEQ * DOUT;
  const __bf16* V = Vt + (size_t)b * DOUT * SEQ;

  // ---- stage Q tile into LDS (16 x 1024 bf16) ----
  {
    const u16x8* src = (const u16x8*)Q;
    u16x8*       dst = (u16x8*)Qlds;
    for (int i = tid; i < 16 * DOUT / 8; i += 128) dst[i] = src[i];
  }
  __syncthreads();

  const f32x8 vzero = {};
  f32x8 o[16];
  #pragma unroll
  for (int c = 0; c < 16; ++c) o[c] = vzero;
  float m_i[8], l_i[8];
  #pragma unroll
  for (int r = 0; r < 8; ++r) { m_i[r] = -INFINITY; l_i[r] = 0.0f; }

  const int nIter = i0 / 128 + 1;        // 128-key super-blocks (causal bound)
  for (int it = 0; it < nIter; ++it) {
    int j0 = it * 128 + w * 32;          // this wave's key slice

    // ---- S = Q * K^T : two 16x16 f32 tiles over K-dim 1024 ----
    f32x8 s0 = vzero, s1 = vzero;
    #pragma unroll 4
    for (int k = 0; k < DOUT; k += 32) {
      bf16x16 a  = load_frag_a_lds(Qlds + k, DOUT, lane);
      bf16x16 b0 = load_frag_b(K + (size_t)j0 * DOUT + k, DOUT, lane);
      bf16x16 b1 = load_frag_b(K + (size_t)(j0 + 16) * DOUT + k, DOUT, lane);
      s0 = wmma_bf16(a, b0, s0);
      s1 = wmma_bf16(a, b1, s1);
    }

    // ---- causal mask, local row-max over this wave's 32 keys ----
    float v0[8], v1[8], mx[8];
    #pragma unroll
    for (int r = 0; r < 8; ++r) {
      int m = 8 * hi + r;
      int i = i0 + m;
      v0[r] = (j0 + lo      <= i) ? s0[r] * KSCALE_LOG2 : -INFINITY;
      v1[r] = (j0 + 16 + lo <= i) ? s1[r] * KSCALE_LOG2 : -INFINITY;
      float t = fmaxf(v0[r], v1[r]);
      #pragma unroll
      for (int off = 1; off < 16; off <<= 1)
        t = fmaxf(t, __shfl_xor(t, off, 32));
      mx[r] = t;
    }
    if (lo == 0) {
      #pragma unroll
      for (int r = 0; r < 8; ++r) statsM[w][8 * hi + r] = mx[r];
    }
    __syncthreads();

    // ---- global max, alpha, P = exp2(S - m), local row-sum -> LDS ----
    float alpha[8];
    #pragma unroll
    for (int r = 0; r < 8; ++r) {
      int m = 8 * hi + r;
      float mb = fmaxf(fmaxf(statsM[0][m], statsM[1][m]),
                       fmaxf(statsM[2][m], statsM[3][m]));
      float mnew = fmaxf(m_i[r], mb);
      alpha[r] = exp2f(m_i[r] - mnew);   // first iter: exp2(-inf) = 0
      m_i[r]   = mnew;
      float p0 = exp2f(v0[r] - mnew);
      float p1 = exp2f(v1[r] - mnew);
      float sum = p0 + p1;
      #pragma unroll
      for (int off = 1; off < 16; off <<= 1)
        sum += __shfl_xor(sum, off, 32);
      if (lo == 0) statsS[w][m] = sum;
      Plds[m * 128 + w * 32 + lo]      = (__bf16)p0;
      Plds[m * 128 + w * 32 + 16 + lo] = (__bf16)p1;
    }
    __syncthreads();

    // ---- accumulate l from all 4 waves' partial sums ----
    #pragma unroll
    for (int r = 0; r < 8; ++r) {
      int m = 8 * hi + r;
      l_i[r] = alpha[r] * l_i[r] +
               (statsS[0][m] + statsS[1][m]) + (statsS[2][m] + statsS[3][m]);
    }

    // ---- O += P * V over this wave's 256-column D slice ----
    bf16x16 pa[4];
    #pragma unroll
    for (int sb = 0; sb < 4; ++sb)
      pa[sb] = load_frag_a_lds(Plds + sb * 32, 128, lane);

    #pragma unroll
    for (int c = 0; c < 16; ++c) {
      int e0 = w * 256 + c * 16;
      #pragma unroll
      for (int r = 0; r < 8; ++r) o[c][r] *= alpha[r];
      #pragma unroll
      for (int sb = 0; sb < 4; ++sb) {
        bf16x16 bv = load_frag_b(V + (size_t)e0 * SEQ + it * 128 + sb * 32,
                                 SEQ, lane);
        o[c] = wmma_bf16(pa[sb], bv, o[c]);
      }
    }
    __syncthreads();                     // Plds/stats reused next iteration
  }

  // ---- epilogue: normalize and store fp32 ----
  float rl[8];
  #pragma unroll
  for (int r = 0; r < 8; ++r) rl[r] = 1.0f / l_i[r];
  #pragma unroll
  for (int c = 0; c < 16; ++c) {
    #pragma unroll
    for (int r = 0; r < 8; ++r) {
      int m = 8 * hi + r;
      out[((size_t)b * SEQ + i0 + m) * DOUT + w * 256 + c * 16 + lo] =
          o[c][r] * rl[r];
    }
  }
}

// ---------------------------------------------------------------------------
// Host launcher
// ---------------------------------------------------------------------------
extern "C" void kernel_launch(void* const* d_in, const int* in_sizes, int n_in,
                              void* d_out, int out_size, void* d_ws, size_t ws_size,
                              hipStream_t stream) {
  const float* x  = (const float*)d_in[0];
  const float* Wq = (const float*)d_in[1];
  const float* Wk = (const float*)d_in[2];
  const float* Wv = (const float*)d_in[3];
  float* out = (float*)d_out;

  // workspace layout (bf16 buffers)
  const size_t nX = (size_t)BATCH * SEQ * DIN;   // 16.7M
  const size_t nW = (size_t)DOUT * DIN;          // 1M
  char* ws = (char*)d_ws;
  __bf16* xb  = (__bf16*)(ws);                       size_t off = nX * 2;
  __bf16* wqb = (__bf16*)(ws + off);                 off += nW * 2;
  __bf16* wkb = (__bf16*)(ws + off);                 off += nW * 2;
  __bf16* wvb = (__bf16*)(ws + off);                 off += nW * 2;
  __bf16* Qb  = (__bf16*)(ws + off);                 off += nX * 2;
  __bf16* Kb  = (__bf16*)(ws + off);                 off += nX * 2;
  __bf16* Vt  = (__bf16*)(ws + off);                 off += nX * 2;
  (void)ws_size; (void)n_in; (void)in_sizes; (void)out_size;

  // Phase 0: convert
  cvt_f32_to_bf16<<<4096, 256, 0, stream>>>(x,  xb,  (int)nX);
  cvt_f32_to_bf16<<<512,  256, 0, stream>>>(Wq, wqb, (int)nW);
  cvt_f32_to_bf16<<<512,  256, 0, stream>>>(Wk, wkb, (int)nW);
  cvt_f32_to_bf16<<<512,  256, 0, stream>>>(Wv, wvb, (int)nW);

  // Phase 1: projections. tiles = (BATCH*SEQ/16) * (DOUT/16) = 65536; /4 waves
  const int projBlocks = (BATCH * SEQ / 16) * (DOUT / 16) / 4;  // 16384
  proj_gemm<<<projBlocks, 128, 0, stream>>>(xb, wqb, Qb, nullptr, 0);
  proj_gemm<<<projBlocks, 128, 0, stream>>>(xb, wkb, Kb, nullptr, 0);
  proj_gemm<<<projBlocks, 128, 0, stream>>>(xb, wvb, nullptr, Vt, 1);

  // Phase 2: attention. One 128-thread block per 16-row query tile.
  attn_fwd<<<BATCH * (SEQ / 16), 128, 0, stream>>>(Qb, Kb, Vt, out);
}
